// MoE_layer_48473000902676
// MI455X (gfx1250) — matrix operand order repaired and checked by reading, not compiled, not run on hardware
//
#include <hip/hip_runtime.h>
#include <hip/hip_bf16.h>
#include <math.h>

typedef unsigned short u16;
typedef unsigned int   u32;

#define NTOK 16384
#define DDIM 512
#define MDIM 1024
#define NEXP 8
#define TILE_N 64
#define TILES_PER_EXP (NTOK / TILE_N)   // 256 (worst case: one expert gets all tokens)

// padded LDS row strides (in bf16 elements) -> 1040B / 2064B rows:
// bank advance of 4 dwords per row => 16 consecutive rows hit 16 distinct
// 4-bank groups of the 64-bank LDS => conflict-free ds_load_b128 A-operand reads.
#define XS_STRIDE 520    // 512 + 8
#define HS_STRIDE 1032   // 1024 + 8

typedef __attribute__((ext_vector_type(16))) __bf16 bf16x16;
typedef __attribute__((ext_vector_type(8)))  __bf16 bf16x8;
typedef __attribute__((ext_vector_type(8)))  float  f32x8;
typedef __attribute__((ext_vector_type(4)))  int    v4i;

__device__ __forceinline__ u16 f2bf(float f) {
    u32 u = __float_as_uint(f);
    u += 0x7FFFu + ((u >> 16) & 1u);      // round-to-nearest-even
    return (u16)(u >> 16);
}

// ---------------- init: zero routing counters ----------------
__global__ void init_kernel(float* imp, int* cnt) {
    int t = threadIdx.x;
    if (t < NEXP) { imp[t] = 0.f; cnt[t] = 0; }
}

// ---------------- x fp32 -> bf16 ----------------
__global__ void cvt_x_kernel(const float* __restrict__ x, u16* __restrict__ xbf) {
    size_t tid = (size_t)blockIdx.x * blockDim.x + threadIdx.x;
    size_t base = tid * 8;
    if (base >= (size_t)NTOK * DDIM) return;
    float4 a = *(const float4*)(x + base);
    float4 b = *(const float4*)(x + base + 4);
    uint4 o;
    o.x = (u32)f2bf(a.x) | ((u32)f2bf(a.y) << 16);
    o.y = (u32)f2bf(a.z) | ((u32)f2bf(a.w) << 16);
    o.z = (u32)f2bf(b.x) | ((u32)f2bf(b.y) << 16);
    o.w = (u32)f2bf(b.z) | ((u32)f2bf(b.w) << 16);
    *(uint4*)(xbf + base) = o;
}

// ------- tiled transpose + convert: src [E][R][C] f32 -> dst [E][C][R] bf16 -------
__global__ void transpose_bf_kernel(const float* __restrict__ src, u16* __restrict__ dst,
                                    int R, int C) {
    __shared__ float tile[32][33];
    int e  = blockIdx.z;
    int c0 = blockIdx.x * 32, r0 = blockIdx.y * 32;
    int tx = threadIdx.x, ty = threadIdx.y;          // 32 x 8
    const float* s = src + (size_t)e * R * C;
    u16*         d = dst + (size_t)e * R * C;
    #pragma unroll
    for (int i = 0; i < 4; ++i)
        tile[ty + i * 8][tx] = s[(size_t)(r0 + ty + i * 8) * C + (c0 + tx)];
    __syncthreads();
    #pragma unroll
    for (int i = 0; i < 4; ++i)
        d[(size_t)(c0 + ty + i * 8) * R + (r0 + tx)] = f2bf(tile[tx][ty + i * 8]);
}

// ---------------- gating: logits, top-2 softmax, routing lists, stats ----------------
__global__ void gate_kernel(const float* __restrict__ x, const float* __restrict__ wg,
                            int* __restrict__ list, float* __restrict__ ga,
                            float* __restrict__ imp, int* __restrict__ cnt) {
    int n = blockIdx.x * blockDim.x + threadIdx.x;
    if (n >= NTOK) return;
    float acc[NEXP];
    #pragma unroll
    for (int e = 0; e < NEXP; ++e) acc[e] = 0.f;
    const float4* xr = (const float4*)(x + (size_t)n * DDIM);
    for (int d4 = 0; d4 < DDIM / 4; ++d4) {
        float4 xv = xr[d4];
        const float* w0 = wg + (size_t)(d4 * 4) * NEXP;   // w_gate is [D][E]
        #pragma unroll
        for (int e = 0; e < NEXP; ++e)
            acc[e] += xv.x * w0[e] + xv.y * w0[NEXP + e] +
                      xv.z * w0[2 * NEXP + e] + xv.w * w0[3 * NEXP + e];
    }
    int i1 = 0; float v1 = acc[0];
    #pragma unroll
    for (int e = 1; e < NEXP; ++e) if (acc[e] > v1) { v1 = acc[e]; i1 = e; }
    int i2 = (i1 == 0) ? 1 : 0; float v2 = acc[i2];
    #pragma unroll
    for (int e = 0; e < NEXP; ++e) if (e != i1 && acc[e] > v2) { v2 = acc[e]; i2 = e; }
    float r  = __expf(v2 - v1);          // softmax over the two kept logits
    float g1 = 1.f / (1.f + r);
    float g2 = r / (1.f + r);
    int p1 = atomicAdd(&cnt[i1], 1);
    list[i1 * NTOK + p1] = (n << 1);
    int p2 = atomicAdd(&cnt[i2], 1);
    list[i2 * NTOK + p2] = (n << 1) | 1;
    ga[n * 2 + 0] = g1;
    ga[n * 2 + 1] = g2;
    atomicAdd(&imp[i1], g1);
    atomicAdd(&imp[i2], g2);
}

// ---------------- fused expert FFN: gather -> GEMM1+GELU -> GEMM2 -> scatter ----------------
__global__ __launch_bounds__(256) void ffn_kernel(
    const u16* __restrict__ xbf, const u16* __restrict__ w1t, const u16* __restrict__ w2t,
    const float* __restrict__ b1, const float* __restrict__ b2,
    const int* __restrict__ list, const float* __restrict__ ga,
    const int* __restrict__ cnt, float* __restrict__ buf) {

    int bid = blockIdx.x;
    int e   = bid / TILES_PER_EXP;
    int t   = bid % TILES_PER_EXP;
    int ce  = cnt[e];                 // uniform across block
    int base = t * TILE_N;
    if (base >= ce) return;           // uniform early exit
    int nvalid = min(TILE_N, ce - base);

    extern __shared__ char smem[];
    u16*   Xs  = (u16*)smem;                         // 64 x 520 bf16  (~65 KB, padded)
    u16*   Hs  = Xs + TILE_N * XS_STRIDE;            // 64 x 1032 bf16 (~129 KB, padded)
    float* gts = (float*)(Hs + TILE_N * HS_STRIDE);  // 64 gates
    int*   asg = (int*)(gts + TILE_N);               // 64 assignment slots

    int tid = threadIdx.x;
    if (tid < TILE_N) {
        int idx   = min(base + tid, ce - 1);      // duplicate last row for tail
        int entry = list[e * NTOK + idx];
        asg[tid]  = entry;
        gts[tid]  = ga[entry];
    }
    __syncthreads();

    // gather 64 bf16 token rows into LDS (each thread: 256B of one quarter-row)
    {
        int row = tid >> 2;
        int q   = tid & 3;
        int tok = asg[row] >> 1;
        const uint4* src = (const uint4*)(xbf + (size_t)tok * DDIM + q * 128);
        uint4*       dst = (uint4*)(Xs + row * XS_STRIDE + q * 128);
#if defined(__gfx1250__) && \
    __has_builtin(__builtin_amdgcn_global_load_async_to_lds_b128) && \
    __has_builtin(__builtin_amdgcn_s_wait_asynccnt)
        // CDNA5 async DMA path: memory -> LDS with no VGPR round trip (ASYNCcnt)
        #pragma unroll
        for (int i = 0; i < 16; ++i) {
            __builtin_amdgcn_global_load_async_to_lds_b128(
                (__attribute__((address_space(1))) v4i*)(src + i),
                (__attribute__((address_space(3))) v4i*)(dst + i),
                0, 0);
        }
        __builtin_amdgcn_s_wait_asynccnt(0);   // our async writes done before the barrier
#else
        #pragma unroll
        for (int i = 0; i < 16; ++i) dst[i] = src[i];
#endif
    }
    __syncthreads();

    int lane = tid & 31;
    int w    = tid >> 5;              // 8 waves
    int rt   = w & 3;                 // row-tile of the 64 output rows
    int hh   = w >> 2;                // column half
    int ln15 = lane & 15;
    int hi16 = (lane >> 4) & 1;       // lanes 16-31 take k 8-15 and 24-31 of each 32-k block

    const u16* w1e = w1t + (size_t)e * MDIM * DDIM;   // [M][D] bf16
    const u16* w2e = w2t + (size_t)e * DDIM * MDIM;   // [D][M] bf16
    const float* b1e = b1 + e * MDIM;
    const float* b2e = b2 + e * DDIM;

    // ---- GEMM1: H = gelu(X(64x512) @ W1(512x1024) + b1) ----
    for (int nt = hh * 32; nt < hh * 32 + 32; ++nt) {
        f32x8 acc = {0.f, 0.f, 0.f, 0.f, 0.f, 0.f, 0.f, 0.f};
        const u16* arow = Xs  + (size_t)(rt * 16 + ln15) * XS_STRIDE + hi16 * 8;
        const u16* bcol = w1e + (size_t)(nt * 16 + ln15) * DDIM + hi16 * 8;
        __builtin_prefetch(bcol + 512, 0, 1);   // next column row (L2-resident weights)
        #pragma unroll
        for (int kb = 0; kb < DDIM / 32; ++kb) {
            bf16x8 a0 = *(const bf16x8*)(arow + kb * 32);
            bf16x8 a1 = *(const bf16x8*)(arow + kb * 32 + 16);
            bf16x8 c0 = *(const bf16x8*)(bcol + kb * 32);
            bf16x8 c1 = *(const bf16x8*)(bcol + kb * 32 + 16);
            bf16x16 A = __builtin_shufflevector(a0, a1, 0,1,2,3,4,5,6,7,8,9,10,11,12,13,14,15);
            bf16x16 B = __builtin_shufflevector(c0, c1, 0,1,2,3,4,5,6,7,8,9,10,11,12,13,14,15);
            acc = __builtin_amdgcn_wmma_f32_16x16x32_bf16(false, A, false, B,
                                                          (short)0, acc, false, false);
        }
        int col   = nt * 16 + ln15;
        float bia = b1e[col];
        #pragma unroll
        for (int j = 0; j < 8; ++j) {
            int row = rt * 16 + j + hi16 * 8;
            float h = acc[j] + bia;
            h = 0.5f * h * (1.f + erff(h * 0.70710678118654752f));   // exact gelu
            Hs[row * HS_STRIDE + col] = f2bf(h);
        }
    }
    __syncthreads();

    // ---- GEMM2: out = (H(64x1024) @ W2(1024x512) + b2) * gate -> scatter ----
    for (int nt = hh * 16; nt < hh * 16 + 16; ++nt) {
        f32x8 acc = {0.f, 0.f, 0.f, 0.f, 0.f, 0.f, 0.f, 0.f};
        const u16* arow = Hs  + (size_t)(rt * 16 + ln15) * HS_STRIDE + hi16 * 8;
        const u16* bcol = w2e + (size_t)(nt * 16 + ln15) * MDIM + hi16 * 8;
        __builtin_prefetch(bcol + 1024, 0, 1);
        #pragma unroll
        for (int kb = 0; kb < MDIM / 32; ++kb) {
            bf16x8 a0 = *(const bf16x8*)(arow + kb * 32);
            bf16x8 a1 = *(const bf16x8*)(arow + kb * 32 + 16);
            bf16x8 c0 = *(const bf16x8*)(bcol + kb * 32);
            bf16x8 c1 = *(const bf16x8*)(bcol + kb * 32 + 16);
            bf16x16 A = __builtin_shufflevector(a0, a1, 0,1,2,3,4,5,6,7,8,9,10,11,12,13,14,15);
            bf16x16 B = __builtin_shufflevector(c0, c1, 0,1,2,3,4,5,6,7,8,9,10,11,12,13,14,15);
            acc = __builtin_amdgcn_wmma_f32_16x16x32_bf16(false, A, false, B,
                                                          (short)0, acc, false, false);
        }
        int col   = nt * 16 + ln15;
        float bia = b2e[col];
        #pragma unroll
        for (int j = 0; j < 8; ++j) {
            int row = rt * 16 + j + hi16 * 8;
            if (row < nvalid) {
                float o = (acc[j] + bia) * gts[row];
                buf[(size_t)asg[row] * DDIM + col] = o;   // per-assignment row, no atomics
            }
        }
    }
}

// ---------------- combine: y[n] = buf[2n] + buf[2n+1] ----------------
__global__ void combine_kernel(const float* __restrict__ buf, float* __restrict__ y) {
    size_t gid  = (size_t)blockIdx.x * blockDim.x + threadIdx.x;
    size_t base = gid * 4;
    if (base >= (size_t)NTOK * DDIM) return;
    int n = (int)(base >> 9);
    int d = (int)(base & 511);
    float4 v0 = *(const float4*)(buf + ((size_t)(2 * n) * DDIM + d));
    float4 v1 = *(const float4*)(buf + ((size_t)(2 * n + 1) * DDIM + d));
    float4 r; r.x = v0.x + v1.x; r.y = v0.y + v1.y; r.z = v0.z + v1.z; r.w = v0.w + v1.w;
    *(float4*)(y + base) = r;
}

// ---------------- CV^2 load-balancing loss ----------------
__global__ void loss_kernel(const float* __restrict__ imp, const int* __restrict__ cnt,
                            float* __restrict__ out) {
    if (threadIdx.x != 0 || blockIdx.x != 0) return;
    float m1 = 0.f, m2 = 0.f;
    for (int e = 0; e < NEXP; ++e) { m1 += imp[e]; m2 += (float)cnt[e]; }
    m1 /= NEXP; m2 /= NEXP;
    float v1 = 0.f, v2 = 0.f;
    for (int e = 0; e < NEXP; ++e) {
        float a = imp[e] - m1; v1 += a * a;
        float b = (float)cnt[e] - m2; v2 += b * b;
    }
    v1 /= (NEXP - 1); v2 /= (NEXP - 1);
    float loss = 0.01f * (v1 / (m1 * m1 + 1e-10f) + v2 / (m2 * m2 + 1e-10f));
    out[(size_t)NTOK * DDIM] = loss;
}

extern "C" void kernel_launch(void* const* d_in, const int* in_sizes, int n_in,
                              void* d_out, int out_size, void* d_ws, size_t ws_size,
                              hipStream_t stream) {
    (void)in_sizes; (void)n_in; (void)out_size; (void)ws_size;
    const float* x  = (const float*)d_in[0];
    const float* wg = (const float*)d_in[1];
    const float* W1 = (const float*)d_in[2];
    const float* b1 = (const float*)d_in[3];
    const float* W2 = (const float*)d_in[4];
    const float* b2 = (const float*)d_in[5];
    float* out = (float*)d_out;

    char* ws = (char*)d_ws;
    size_t off = 0;
    auto carve = [&](size_t bytes) -> char* {
        char* p = ws + off;
        off = (off + bytes + 255) & ~(size_t)255;
        return p;
    };
    u16*   xbf  = (u16*)  carve((size_t)NTOK * DDIM * 2);          // 16 MB
    u16*   w1t  = (u16*)  carve((size_t)NEXP * MDIM * DDIM * 2);   //  8 MB
    u16*   w2t  = (u16*)  carve((size_t)NEXP * DDIM * MDIM * 2);   //  8 MB
    float* buf  = (float*)carve((size_t)2 * NTOK * DDIM * 4);      // 64 MB
    int*   list = (int*)  carve((size_t)NEXP * NTOK * 4);
    float* ga   = (float*)carve((size_t)2 * NTOK * 4);
    float* imp  = (float*)carve(NEXP * 4);
    int*   cnt  = (int*)  carve(NEXP * 4);

    init_kernel<<<1, 32, 0, stream>>>(imp, cnt);
    cvt_x_kernel<<<(NTOK * DDIM / 8 + 255) / 256, 256, 0, stream>>>(x, xbf);
    dim3 tb(32, 8);
    transpose_bf_kernel<<<dim3(MDIM / 32, DDIM / 32, NEXP), tb, 0, stream>>>(W1, w1t, DDIM, MDIM);
    transpose_bf_kernel<<<dim3(DDIM / 32, MDIM / 32, NEXP), tb, 0, stream>>>(W2, w2t, MDIM, DDIM);
    gate_kernel<<<NTOK / 256, 256, 0, stream>>>(x, wg, list, ga, imp, cnt);

    size_t shmem = (size_t)TILE_N * XS_STRIDE * 2 + (size_t)TILE_N * HS_STRIDE * 2
                 + TILE_N * sizeof(float) + TILE_N * sizeof(int);  // ~195 KB of 320 KB WGP LDS
    ffn_kernel<<<NEXP * TILES_PER_EXP, 256, shmem, stream>>>(xbf, w1t, w2t, b1, b2,
                                                             list, ga, cnt, buf);
    combine_kernel<<<(NTOK * DDIM / 4 + 255) / 256, 256, 0, stream>>>(buf, out);
    loss_kernel<<<1, 32, 0, stream>>>(imp, cnt, out);
}
